// ResidualAttentionBlock_206158430844
// MI455X (gfx1250) — matrix operand order repaired and checked by reading, not compiled, not run on hardware
//
#include <hip/hip_runtime.h>
#include <hip/hip_bf16.h>

#define D_MODEL 768
#define N_HEAD  12
#define SEQ     2048
#define BATCH   4
#define DH      64
#define NHEADS_TOTAL (N_HEAD * BATCH)   // 48
#define MROWS   (SEQ * BATCH)           // 8192
#define FFN     (4 * D_MODEL)           // 3072
#define QKVN    (3 * D_MODEL)           // 2304

typedef __bf16 bf16_t;
typedef __attribute__((ext_vector_type(16))) __bf16 v16bf;
typedef __attribute__((ext_vector_type(8)))  __bf16 v8bf;
typedef __attribute__((ext_vector_type(8)))  float  v8f;

union BF16x16 { v16bf v; v8bf h[2]; };

__device__ __forceinline__ v8bf load8bf(const bf16_t* p) {
  return *reinterpret_cast<const v8bf*>(p);
}
// A-fragment (16x32 bf16): lane = row (lane&15); elements K = 16*half + 0..15 contiguous.
// p must already point at (row, k0 + 16*half).
__device__ __forceinline__ v16bf loadA(const bf16_t* p) {
  BF16x16 u; u.h[0] = load8bf(p); u.h[1] = load8bf(p + 8); return u.v;
}
// B-fragment (32x16 bf16): lane = col (lane&15); elements K = 16*half + 0..15 contiguous.
__device__ __forceinline__ v16bf loadB(const bf16_t* p) {
  return *reinterpret_cast<const v16bf*>(p);
}
__device__ __forceinline__ v8f wmma_bf16(v16bf a, v16bf b, v8f c) {
  return __builtin_amdgcn_wmma_f32_16x16x32_bf16(false, a, false, b, (short)0, c,
                                                 false, false);
}

// ---------------------------------------------------------------- fp32 -> bf16
__global__ void f32_to_bf16_kernel(const float* __restrict__ in,
                                   bf16_t* __restrict__ out, int n) {
  int i = blockIdx.x * blockDim.x + threadIdx.x;
  int stride = gridDim.x * blockDim.x;
  for (; i < n; i += stride) out[i] = (bf16_t)in[i];
}

// ---------------------------------------------------------------- LayerNorm
// One 256-thread block per row of 768; fp32 stats, bf16 output.
__global__ __launch_bounds__(256) void ln_bf16_kernel(
    const float* __restrict__ X, const float* __restrict__ w,
    const float* __restrict__ b, bf16_t* __restrict__ Y) {
  int row = blockIdx.x;
  const float* xr = X + (size_t)row * D_MODEL;
  float v[3], s = 0.f, ss = 0.f;
#pragma unroll
  for (int i = 0; i < 3; ++i) {
    v[i] = xr[threadIdx.x + 256 * i];
    s += v[i]; ss += v[i] * v[i];
  }
#pragma unroll
  for (int m = 16; m > 0; m >>= 1) {
    s  += __shfl_xor(s,  m);
    ss += __shfl_xor(ss, m);
  }
  __shared__ float red[2][8];
  int wv = threadIdx.x >> 5, lane = threadIdx.x & 31;
  if (lane == 0) { red[0][wv] = s; red[1][wv] = ss; }
  __syncthreads();
  float ts = 0.f, tss = 0.f;
#pragma unroll
  for (int i = 0; i < 8; ++i) { ts += red[0][i]; tss += red[1][i]; }
  float mean = ts * (1.f / D_MODEL);
  float var  = tss * (1.f / D_MODEL) - mean * mean;
  float rstd = rsqrtf(var + 1e-5f);
  bf16_t* yr = Y + (size_t)row * D_MODEL;
#pragma unroll
  for (int i = 0; i < 3; ++i) {
    int c = threadIdx.x + 256 * i;
    yr[c] = (bf16_t)((v[i] - mean) * rstd * w[c] + b[c]);
  }
}

// ---------------------------------------------------------------- GEMM
// C[m,c] = sum_k A[m,k] * W[c,k]  (+ epilogue).  A,W bf16 row-major.
// Wave computes 64x32; workgroup = 8 waves (2x4) => 128x128 tile.
// MODE 0: qkv scatter (+bias) -> Qb,Kb [N,H,L,Dh], Vt [N,H,Dh,L]
// MODE 1: outF = val + bias + resid
// MODE 2: outB = bf16(QuickGELU(val + bias))
// MODE 3: outF = val + bias + resid   (final projection -> d_out)
template <int MODE>
__global__ __launch_bounds__(256) void gemm_bf16_kernel(
    const bf16_t* __restrict__ A, const bf16_t* __restrict__ W,
    const float* __restrict__ bias, const float* __restrict__ resid,
    float* __restrict__ outF, bf16_t* __restrict__ outB,
    bf16_t* __restrict__ Qb, bf16_t* __restrict__ Kb, bf16_t* __restrict__ Vt,
    int Nout, int K) {
  int lane = threadIdx.x & 31;
  int wv   = threadIdx.x >> 5;
  int wm = wv >> 2, wn = wv & 3;
  int half = lane >> 4, ln = lane & 15;
  int mbase = blockIdx.y * 128 + wm * 64;
  int nbase = blockIdx.x * 128 + wn * 32;

  v8f acc[4][2] = {};
  for (int k0 = 0; k0 < K; k0 += 32) {
    // Branchless look-ahead prefetch (global_prefetch_b8): clamp keeps the
    // address in-bounds and the loop body unsplit.
    int kp = k0 + 64; if (kp > K - 32) kp = K - 32;
    const bf16_t* pa = A + (size_t)(mbase + lane) * K + kp;
    __builtin_prefetch(pa, 0, 3);
    __builtin_prefetch(pa + (size_t)32 * K, 0, 3);
    __builtin_prefetch(W + (size_t)(nbase + lane) * K + kp, 0, 3);

    // Hoist all fragments into distinct locals so the loads clause together
    // and WMMA waits stagger instead of serializing.
    v16bf a[4];
#pragma unroll
    for (int i = 0; i < 4; ++i)
      a[i] = loadA(A + (size_t)(mbase + i * 16 + ln) * K + k0 + 16 * half);
    v16bf bfr[2];
#pragma unroll
    for (int j = 0; j < 2; ++j)
      bfr[j] = loadB(W + (size_t)(nbase + j * 16 + ln) * K + k0 + 16 * half);
#pragma unroll
    for (int i = 0; i < 4; ++i)
#pragma unroll
      for (int j = 0; j < 2; ++j)
        acc[i][j] = wmma_bf16(a[i], bfr[j], acc[i][j]);
  }

#pragma unroll
  for (int i = 0; i < 4; ++i) {
#pragma unroll
    for (int j = 0; j < 2; ++j) {
      int c = nbase + j * 16 + ln;
      float bc = bias[c];
#pragma unroll
      for (int r = 0; r < 8; ++r) {
        int m = mbase + i * 16 + r + 8 * half;
        float val = acc[i][j][r] + bc;
        if (MODE == 0) {
          int g  = c / D_MODEL;
          int cc = c - g * D_MODEL;
          int hh = cc >> 6, d = cc & 63;
          int l = m >> 2, nb = m & 3;
          size_t nhh = (size_t)(nb * N_HEAD + hh);
          if (g == 0)       Qb[(nhh * SEQ + l) * DH + d] = (bf16_t)val;
          else if (g == 1)  Kb[(nhh * SEQ + l) * DH + d] = (bf16_t)val;
          else              Vt[(nhh * DH + d) * SEQ + l] = (bf16_t)val;
        } else if (MODE == 1 || MODE == 3) {
          size_t idx = (size_t)m * Nout + c;
          outF[idx] = val + resid[idx];
        } else {  // MODE == 2: QuickGELU
          float g = val * (1.f / (1.f + __expf(-1.702f * val)));
          outB[(size_t)m * Nout + c] = (bf16_t)g;
        }
      }
    }
  }
}

// ---------------------------------------------------------------- Flash attention
// grid = (SEQ/64, 48), block = 128 (4 waves).  Each wave: 16 query rows,
// streaming 32 keys per iteration with online softmax.
__global__ __launch_bounds__(128) void flash_attn_kernel(
    const bf16_t* __restrict__ Q, const bf16_t* __restrict__ K,
    const bf16_t* __restrict__ Vt, bf16_t* __restrict__ O) {
  __shared__ __align__(16) bf16_t lds[4][16 * 32];
  int tid = threadIdx.x;
  int wv = tid >> 5, lane = tid & 31;
  int half = lane >> 4, ln = lane & 15;
  int nh = blockIdx.y;
  int nb = nh / N_HEAD, hh = nh - nb * N_HEAD;
  int qbase = blockIdx.x * 64 + wv * 16;

  const bf16_t* Qh = Q  + (size_t)nh * SEQ * DH;
  const bf16_t* Kh = K  + (size_t)nh * SEQ * DH;
  const bf16_t* Vh = Vt + (size_t)nh * DH * SEQ;

  v16bf aQ[2];
#pragma unroll
  for (int t = 0; t < 2; ++t)
    aQ[t] = loadA(Qh + (size_t)(qbase + ln) * DH + t * 32 + 16 * half);

  float mrow[8], lrow[8];
  v8f acc[4] = {};
#pragma unroll
  for (int r = 0; r < 8; ++r) { mrow[r] = -1e30f; lrow[r] = 0.f; }
  const float scale = 0.125f;  // 1/sqrt(64)

  for (int j0 = 0; j0 < SEQ; j0 += 32) {
    // Branchless look-ahead prefetch of next key rows / V columns.
    int jp = j0 + 32; if (jp > SEQ - 32) jp = SEQ - 32;
    __builtin_prefetch(Kh + (size_t)(jp + lane) * DH, 0, 3);
    __builtin_prefetch(Vh + (size_t)lane * SEQ + jp, 0, 3);
    __builtin_prefetch(Vh + (size_t)(lane + 32) * SEQ + jp, 0, 3);

    // All four K B-fragments in distinct locals, then interleave the two
    // independent accumulate chains (s0, s1).
    const bf16_t* kp0 = Kh + (size_t)(j0 + ln) * DH + 16 * half;
    const bf16_t* kp1 = Kh + (size_t)(j0 + 16 + ln) * DH + 16 * half;
    v16bf bK00 = loadB(kp0);
    v16bf bK01 = loadB(kp0 + 32);
    v16bf bK10 = loadB(kp1);
    v16bf bK11 = loadB(kp1 + 32);
    v8f s0 = {}, s1 = {};
    s0 = wmma_bf16(aQ[0], bK00, s0);
    s1 = wmma_bf16(aQ[0], bK10, s1);
    s0 = wmma_bf16(aQ[1], bK01, s0);
    s1 = wmma_bf16(aQ[1], bK11, s1);

    float p0[8], p1[8], mt[8], rs[8], alpha[8];
#pragma unroll
    for (int r = 0; r < 8; ++r) {
      p0[r] = s0[r] * scale;
      p1[r] = s1[r] * scale;
      mt[r] = fmaxf(p0[r], p1[r]);
    }
#pragma unroll
    for (int m = 1; m <= 8; m <<= 1)
#pragma unroll
      for (int r = 0; r < 8; ++r) mt[r] = fmaxf(mt[r], __shfl_xor(mt[r], m));
#pragma unroll
    for (int r = 0; r < 8; ++r) {
      float mn = fmaxf(mrow[r], mt[r]);
      alpha[r] = __expf(mrow[r] - mn);
      mrow[r] = mn;
      p0[r] = __expf(p0[r] - mn);
      p1[r] = __expf(p1[r] - mn);
      rs[r] = p0[r] + p1[r];
    }
#pragma unroll
    for (int m = 1; m <= 8; m <<= 1)
#pragma unroll
      for (int r = 0; r < 8; ++r) rs[r] += __shfl_xor(rs[r], m);
#pragma unroll
    for (int r = 0; r < 8; ++r) lrow[r] = lrow[r] * alpha[r] + rs[r];
#pragma unroll
    for (int t = 0; t < 4; ++t)
#pragma unroll
      for (int r = 0; r < 8; ++r) acc[t][r] *= alpha[r];

    // C-layout -> A-fragment re-layout through LDS (1KB per wave).
#pragma unroll
    for (int r = 0; r < 8; ++r) {
      int row = r + 8 * half;
      lds[wv][row * 32 + ln]      = (bf16_t)p0[r];
      lds[wv][row * 32 + 16 + ln] = (bf16_t)p1[r];
    }
    __syncthreads();
    v16bf aP = loadA(&lds[wv][ln * 32 + 16 * half]);
    v16bf bV[4];
#pragma unroll
    for (int t = 0; t < 4; ++t)
      bV[t] = loadB(Vh + (size_t)(t * 16 + ln) * SEQ + j0 + 16 * half);
#pragma unroll
    for (int t = 0; t < 4; ++t)
      acc[t] = wmma_bf16(aP, bV[t], acc[t]);
    __syncthreads();
  }

  float inv[8];
#pragma unroll
  for (int r = 0; r < 8; ++r) inv[r] = 1.f / lrow[r];
#pragma unroll
  for (int t = 0; t < 4; ++t) {
#pragma unroll
    for (int r = 0; r < 8; ++r) {
      int l = qbase + r + 8 * half;
      int col = hh * DH + t * 16 + ln;
      O[((size_t)l * BATCH + nb) * D_MODEL + col] = (bf16_t)(acc[t][r] * inv[r]);
    }
  }
}

// ---------------------------------------------------------------- launch
extern "C" void kernel_launch(void* const* d_in, const int* in_sizes, int n_in,
                              void* d_out, int out_size, void* d_ws, size_t ws_size,
                              hipStream_t stream) {
  (void)in_sizes; (void)n_in; (void)out_size; (void)ws_size;
  const float* x      = (const float*)d_in[0];
  const float* ln1_w  = (const float*)d_in[1];
  const float* ln1_b  = (const float*)d_in[2];
  const float* w_qkv  = (const float*)d_in[3];
  const float* b_qkv  = (const float*)d_in[4];
  const float* w_out  = (const float*)d_in[5];
  const float* b_out  = (const float*)d_in[6];
  const float* ln2_w  = (const float*)d_in[7];
  const float* ln2_b  = (const float*)d_in[8];
  const float* w_fc   = (const float*)d_in[9];
  const float* b_fc   = (const float*)d_in[10];
  const float* w_proj = (const float*)d_in[11];
  const float* b_proj = (const float*)d_in[12];

  char* ws = (char*)d_ws;
  size_t off = 0;
  auto alloc = [&](size_t bytes) {
    size_t o = off;
    off = (off + bytes + 255) & ~(size_t)255;
    return o;
  };
  bf16_t* wqkv_b  = (bf16_t*)(ws + alloc((size_t)QKVN * D_MODEL * 2));
  bf16_t* wout_b  = (bf16_t*)(ws + alloc((size_t)D_MODEL * D_MODEL * 2));
  bf16_t* wfc_b   = (bf16_t*)(ws + alloc((size_t)FFN * D_MODEL * 2));
  bf16_t* wproj_b = (bf16_t*)(ws + alloc((size_t)D_MODEL * FFN * 2));
  bf16_t* H1      = (bf16_t*)(ws + alloc((size_t)MROWS * D_MODEL * 2));
  bf16_t* Qb      = (bf16_t*)(ws + alloc((size_t)NHEADS_TOTAL * SEQ * DH * 2));
  bf16_t* Kb      = (bf16_t*)(ws + alloc((size_t)NHEADS_TOTAL * SEQ * DH * 2));
  bf16_t* Vt      = (bf16_t*)(ws + alloc((size_t)NHEADS_TOTAL * DH * SEQ * 2));
  bf16_t* Ao      = (bf16_t*)(ws + alloc((size_t)MROWS * D_MODEL * 2));
  float*  x1      = (float*) (ws + alloc((size_t)MROWS * D_MODEL * 4));
  bf16_t* H2      = (bf16_t*)(ws + alloc((size_t)MROWS * D_MODEL * 2));
  bf16_t* F       = (bf16_t*)(ws + alloc((size_t)MROWS * FFN * 2));

  // 1) weights fp32 -> bf16
  {
    int n;
    n = QKVN * D_MODEL;   f32_to_bf16_kernel<<<(n + 255) / 256, 256, 0, stream>>>(w_qkv,  wqkv_b,  n);
    n = D_MODEL * D_MODEL;f32_to_bf16_kernel<<<(n + 255) / 256, 256, 0, stream>>>(w_out,  wout_b,  n);
    n = FFN * D_MODEL;    f32_to_bf16_kernel<<<(n + 255) / 256, 256, 0, stream>>>(w_fc,   wfc_b,   n);
    n = D_MODEL * FFN;    f32_to_bf16_kernel<<<(n + 255) / 256, 256, 0, stream>>>(w_proj, wproj_b, n);
  }
  // 2) ln1
  ln_bf16_kernel<<<MROWS, 256, 0, stream>>>(x, ln1_w, ln1_b, H1);
  // 3) qkv GEMM + scatter
  gemm_bf16_kernel<0><<<dim3(QKVN / 128, MROWS / 128), 256, 0, stream>>>(
      H1, wqkv_b, b_qkv, nullptr, nullptr, nullptr, Qb, Kb, Vt, QKVN, D_MODEL);
  // 4) flash attention
  flash_attn_kernel<<<dim3(SEQ / 64, NHEADS_TOTAL), 128, 0, stream>>>(Qb, Kb, Vt, Ao);
  // 5) out-proj + residual -> x1
  gemm_bf16_kernel<1><<<dim3(D_MODEL / 128, MROWS / 128), 256, 0, stream>>>(
      Ao, wout_b, b_out, x, x1, nullptr, nullptr, nullptr, nullptr, D_MODEL, D_MODEL);
  // 6) ln2
  ln_bf16_kernel<<<MROWS, 256, 0, stream>>>(x1, ln2_w, ln2_b, H2);
  // 7) fc + QuickGELU
  gemm_bf16_kernel<2><<<dim3(FFN / 128, MROWS / 128), 256, 0, stream>>>(
      H2, wfc_b, b_fc, nullptr, nullptr, F, nullptr, nullptr, nullptr, FFN, D_MODEL);
  // 8) proj + residual -> d_out
  gemm_bf16_kernel<3><<<dim3(D_MODEL / 128, MROWS / 128), 256, 0, stream>>>(
      F, wproj_b, b_proj, x1, (float*)d_out, nullptr, nullptr, nullptr, nullptr,
      D_MODEL, FFN);
}
